// TransformerPolicy_18880676233690
// MI455X (gfx1250) — compile-verified, matrix operand order checked by reading, not run on hardware
//
// MI455X (gfx1250) ViT policy forward: bf16 WMMA GEMMs + LN/softmax epilogues.
// Double-buffered LDS pipeline; async global->LDS staging (ASYNCcnt) where legal.
#include <hip/hip_runtime.h>

typedef __attribute__((ext_vector_type(16))) __bf16 v16bf;
typedef __attribute__((ext_vector_type(8)))  float  v8f;
typedef __attribute__((ext_vector_type(4)))  int    v4i;

static constexpr int BATCH = 32, TOK = 196, EMB = 768, NHEAD = 12, DH = 64;
static constexpr int FFDIM = 3072, NBLK = 8, TFD = 512;
static constexpr int ROWS  = BATCH * TOK;      // 6272
static constexpr long long FLATD = (long long)TOK * EMB; // 150528
static constexpr int LD_S = 208;               // padded score cols (13*16)
static constexpr int LD_P = 224;               // padded P cols (7*32)

__device__ __forceinline__ __bf16 f2bf(float f) {
  unsigned int u = __builtin_bit_cast(unsigned int, f);
  unsigned int r = u + 0x7FFFu + ((u >> 16) & 1u);
  unsigned short h = (unsigned short)(r >> 16);
  return __builtin_bit_cast(__bf16, h);
}

// 16B/lane global->LDS copy. Uses the CDNA5 async-to-LDS path when the
// toolchain exposes it (ASYNCcnt-tracked, no VGPR round trip).
__device__ __forceinline__ void async_cp16(const void* g, void* l) {
#if __has_builtin(__builtin_amdgcn_global_load_async_to_lds_b128)
  __builtin_amdgcn_global_load_async_to_lds_b128(
      (__attribute__((address_space(1))) v4i*)const_cast<void*>(g),
      (__attribute__((address_space(3))) v4i*)l, 0, 0);
#else
  *(uint4*)l = *(const uint4*)g;
#endif
}

__device__ __forceinline__ void async_wait_all() {
#if __has_builtin(__builtin_amdgcn_s_wait_asynccnt)
  __builtin_amdgcn_s_wait_asynccnt(0);
#else
  asm volatile("s_wait_asynccnt 0" ::: "memory");
#endif
}

__device__ __forceinline__ v16bf frag_ld(const __bf16* p) {
  // lane fragment: 8 bf16 at K=kb.. plus 8 bf16 at K=kb+16.. (ISA 7.12.2 layout)
  v16bf f;
  uint4* d = reinterpret_cast<uint4*>(&f);
  d[0] = *reinterpret_cast<const uint4*>(p);
  d[1] = *reinterpret_cast<const uint4*>(p + 16);
  return f;
}

// C[M,N](f32) = A[M,K](bf16,row-major) x B
//   BT=1 : B given as [N,K] row-major (i.e. B^T), dtype bf16 (BF16B=1) or f32
//   BT=0 : B given as [K,N] row-major f32, transposed into LDS
//   ACC  : C += result (residual accumulate)
//   SPLITK: blockIdx.z selects K-chunk of kChunk; partials written at C + z*sCo
template<bool BT, bool BF16B, bool ACC, bool SPLITK>
__global__ __launch_bounds__(256) void gemm_wmma_bf16(
    const __bf16* __restrict__ A, const void* __restrict__ Bv,
    float* __restrict__ C, const float* __restrict__ bias,
    int M, int N, int K, int lda, int ldb, int ldc,
    long long sAo, long long sAi, long long sBo, long long sBi,
    long long sCo, long long sCi, int inner, int kChunk)
{
  __shared__ __bf16 As[2][128][40];  // [buf][m][k], stride 40 avoids conflicts
  __shared__ __bf16 Bs[2][128][40];  // [buf][n][k]

  const int tid = threadIdx.x;
  const int m0 = blockIdx.y * 128;
  const int n0 = blockIdx.x * 128;

  int kBeg = 0, kEnd = K;
  if constexpr (SPLITK) {
    kBeg = blockIdx.z * kChunk;
    kEnd = kBeg + kChunk; if (kEnd > K) kEnd = K;
    if (kBeg >= K) return;
    C += (long long)blockIdx.z * sCo;
  } else {
    const int z  = blockIdx.z;
    const int zo = z / inner, zi = z - zo * inner;
    A  += zo * sAo + zi * sAi;
    C  += zo * sCo + zi * sCi;
    Bv  = (const void*)((const char*)Bv + (zo * sBo + zi * sBi) * (BF16B ? 2 : 4));
  }
  const float*  Bf = (const float*)Bv;
  const __bf16* Bh = (const __bf16*)Bv;

  const int lane = tid & 31;
  const int wv   = tid >> 5;            // 8 wave32 waves
  const int wm   = (wv >> 1) << 5;      // 0,32,64,96
  const int wn   = (wv & 1) << 6;       // 0,64
  const int lm   = lane & 15;
  const int kb   = (lane < 16) ? 0 : 8; // half-wave K base (ISA layout)
  const __bf16 bz = f2bf(0.f);

  auto stageA = [&](int k0, int buf) {
    const bool full = (m0 + 127 < M) && (k0 + 31 < K);  // uniform
    #pragma unroll
    for (int c = 0; c < 2; ++c) {
      int id = tid + (c << 8);
      int r  = id >> 2;
      int ck = (id & 3) << 3;
      int gm = m0 + r, gk = k0 + ck;
      if (full) {
        async_cp16(A + (long long)gm * lda + gk, &As[buf][r][ck]);
      } else {
        int cm = gm < M ? gm : M - 1;
        const __bf16* src = A + (long long)cm * lda;
        __bf16 t[8];
        #pragma unroll
        for (int j = 0; j < 8; ++j) {
          int gkj = gk + j;
          int ckk = gkj < K ? gkj : K - 1;
          __bf16 v0 = src[ckk];                       // clamped, always in-bounds
          t[j] = (gm < M && gkj < K) ? v0 : bz;       // select, no branches
        }
        *(uint4*)(&As[buf][r][ck]) = *(uint4*)t;
      }
    }
  };

  auto stageB = [&](int k0, int buf) {
    if constexpr (BT) {
      const bool full = (n0 + 127 < N) && (k0 + 31 < K);  // uniform
      #pragma unroll
      for (int c = 0; c < 2; ++c) {
        int id = tid + (c << 8);
        int r  = id >> 2;
        int ck = (id & 3) << 3;
        int gn = n0 + r, gk = k0 + ck;
        if constexpr (BF16B) {
          if (full) {
            async_cp16(Bh + (long long)gn * ldb + gk, &Bs[buf][r][ck]);
          } else {
            int cn = gn < N ? gn : N - 1;
            const __bf16* src = Bh + (long long)cn * ldb;
            __bf16 t[8];
            #pragma unroll
            for (int j = 0; j < 8; ++j) {
              int gkj = gk + j;
              int ckk = gkj < K ? gkj : K - 1;
              __bf16 v0 = src[ckk];
              t[j] = (gn < N && gkj < K) ? v0 : bz;
            }
            *(uint4*)(&Bs[buf][r][ck]) = *(uint4*)t;
          }
        } else {
          __bf16 t[8];
          if (full) {
            float4 f0 = *reinterpret_cast<const float4*>(Bf + (long long)gn * ldb + gk);
            float4 f1 = *reinterpret_cast<const float4*>(Bf + (long long)gn * ldb + gk + 4);
            t[0] = f2bf(f0.x); t[1] = f2bf(f0.y); t[2] = f2bf(f0.z); t[3] = f2bf(f0.w);
            t[4] = f2bf(f1.x); t[5] = f2bf(f1.y); t[6] = f2bf(f1.z); t[7] = f2bf(f1.w);
          } else {
            int cn = gn < N ? gn : N - 1;
            const float* src = Bf + (long long)cn * ldb;
            #pragma unroll
            for (int j = 0; j < 8; ++j) {
              int gkj = gk + j;
              int ckk = gkj < K ? gkj : K - 1;
              float v0 = src[ckk];
              t[j] = (gn < N && gkj < K) ? f2bf(v0) : bz;
            }
          }
          *(uint4*)(&Bs[buf][r][ck]) = *(uint4*)t;
        }
      }
    } else {
      // B is [K,N] f32: coalesced read along n, transposed write into LDS
      #pragma unroll
      for (int c = 0; c < 4; ++c) {
        int id = tid + (c << 8);      // 0..1023
        int kk = id >> 5;             // 0..31
        int n4 = (id & 31) << 2;      // 0..124
        int gk = k0 + kk, gn = n0 + n4;
        float f[4];
        bool okk = gk < K;
        if (okk && gn + 3 < N) {
          *(float4*)f = *reinterpret_cast<const float4*>(Bf + (long long)gk * ldb + gn);
          if ((tid & 3) == 0 && gk + 64 < K)   // prefetch 2 tiles ahead
            __builtin_prefetch(Bf + (long long)(gk + 64) * ldb + gn, 0, 1);
        } else {
          int cgk = okk ? gk : K - 1;
          const float* src = Bf + (long long)cgk * ldb;
          #pragma unroll
          for (int j = 0; j < 4; ++j) {
            int gnj = gn + j;
            int cgn = gnj < N ? gnj : N - 1;
            float v0 = src[cgn];
            f[j] = (okk && gnj < N) ? v0 : 0.f;
          }
        }
        #pragma unroll
        for (int j = 0; j < 4; ++j) Bs[buf][n4 + j][kk] = f2bf(f[j]);
      }
    }
  };

  v8f acc[2][4];
  #pragma unroll
  for (int i = 0; i < 2; ++i)
    #pragma unroll
    for (int j = 0; j < 4; ++j)
      #pragma unroll
      for (int e = 0; e < 8; ++e) acc[i][j][e] = 0.f;

  // ---- double-buffered pipeline: one barrier per K-step ----
  int cur = 0;
  stageA(kBeg, 0);
  stageB(kBeg, 0);
  async_wait_all();
  __syncthreads();
  for (int k0 = kBeg; k0 < kEnd; k0 += 32) {
    const int kn = k0 + 32;
    if (kn < kEnd) { stageA(kn, cur ^ 1); stageB(kn, cur ^ 1); }

    // fragments + WMMA (EXEC all-ones: no divergence here)
    v16bf af[2], bfr[4];
    #pragma unroll
    for (int mt = 0; mt < 2; ++mt) af[mt] = frag_ld(&As[cur][wm + mt * 16 + lm][kb]);
    #pragma unroll
    for (int nt = 0; nt < 4; ++nt) bfr[nt] = frag_ld(&Bs[cur][wn + nt * 16 + lm][kb]);
    #pragma unroll
    for (int mt = 0; mt < 2; ++mt)
      #pragma unroll
      for (int nt = 0; nt < 4; ++nt)
        acc[mt][nt] = __builtin_amdgcn_wmma_f32_16x16x32_bf16(
            false, af[mt], false, bfr[nt], (short)0, acc[mt][nt], false, false);

    async_wait_all();   // next-buffer async copies landed (this wave)
    __syncthreads();    // all waves: staging done, current buffer free
    cur ^= 1;
  }

  // ---- epilogue: VGPR r <-> M = r + 8*(lane>=16), N = lane&15 ----
  #pragma unroll
  for (int mt = 0; mt < 2; ++mt)
    #pragma unroll
    for (int nt = 0; nt < 4; ++nt)
      #pragma unroll
      for (int r = 0; r < 8; ++r) {
        int m = m0 + wm + mt * 16 + r + ((lane >> 4) << 3);
        int n = n0 + wn + nt * 16 + (lane & 15);
        if (m < M && n < N) {
          float val = acc[mt][nt][r];
          if (bias) val += bias[n];
          long long idx = (long long)m * ldc + n;
          if constexpr (ACC) C[idx] += val; else C[idx] = val;
        }
      }
}

// LayerNorm over D cols, one workgroup per row.
// MODE 0: out bf16 = LN ; MODE 1: out bf16 = relu(LN) ; MODE 2: out f32 += LN
template<int MODE>
__global__ __launch_bounds__(256) void ln_rows(
    const float* __restrict__ in, const float* __restrict__ addb,
    const float* __restrict__ g, const float* __restrict__ b,
    void* __restrict__ out, int D, int ldin, int ldout)
{
  const int row = blockIdx.x, tid = threadIdx.x;
  const float* x = in + (long long)row * ldin;
  float s = 0.f, q = 0.f;
  for (int j = tid; j < D; j += 256) {
    float v = x[j] + (addb ? addb[j] : 0.f);
    s += v; q += v * v;
  }
  __shared__ float r1[256], r2[256];
  r1[tid] = s; r2[tid] = q;
  __syncthreads();
  for (int off = 128; off > 0; off >>= 1) {
    if (tid < off) { r1[tid] += r1[tid + off]; r2[tid] += r2[tid + off]; }
    __syncthreads();
  }
  float mean = r1[0] / D;
  float var  = r2[0] / D - mean * mean;
  float rstd = rsqrtf(var + 1e-5f);
  if constexpr (MODE == 2) {
    float* o = (float*)out + (long long)row * ldout;
    for (int j = tid; j < D; j += 256) {
      float v = x[j] + (addb ? addb[j] : 0.f);
      o[j] += (v - mean) * rstd * g[j] + b[j];
    }
  } else {
    __bf16* o = (__bf16*)out + (long long)row * ldout;
    for (int j = tid; j < D; j += 256) {
      float v = x[j] + (addb ? addb[j] : 0.f);
      float y = (v - mean) * rstd * g[j] + b[j];
      if (MODE == 1) y = y > 0.f ? y : 0.f;
      o[j] = f2bf(y);
    }
    for (int j = D + tid; j < ldout; j += 256) o[j] = f2bf(0.f);
  }
}

// softmax over `cols` (<=256), write bf16 zero-padded to ldP
__global__ __launch_bounds__(256) void softmax_bf16(
    const float* __restrict__ S, __bf16* __restrict__ P,
    int cols, int ldS, int ldP, float scale)
{
  const long long row = blockIdx.x;
  const int tid = threadIdx.x;
  float v = (tid < cols) ? S[row * ldS + tid] * scale : -3.0e38f;
  __shared__ float red[256];
  red[tid] = v; __syncthreads();
  for (int off = 128; off > 0; off >>= 1) {
    if (tid < off) red[tid] = fmaxf(red[tid], red[tid + off]);
    __syncthreads();
  }
  float mx = red[0];
  __syncthreads();
  float e = (tid < cols) ? __expf(v - mx) : 0.f;
  red[tid] = e; __syncthreads();
  for (int off = 128; off > 0; off >>= 1) {
    if (tid < off) red[tid] += red[tid + off];
    __syncthreads();
  }
  float inv = 1.f / red[0];
  if (tid < ldP) P[row * ldP + tid] = f2bf((tid < cols) ? e * inv : 0.f);
}

// non-overlapping 16x16 patch extraction -> bf16 [B*T, 768]
__global__ __launch_bounds__(256) void im2col_bf16(
    const float* __restrict__ fov, __bf16* __restrict__ Ap)
{
  long long idx = (long long)blockIdx.x * 256 + threadIdx.x;
  if (idx >= (long long)ROWS * 768) return;
  int row = (int)(idx / 768), col = (int)(idx - (long long)row * 768);
  int b = row / TOK, t = row - b * TOK;
  int py = t / 14, px = t - py * 14;
  int c = col >> 8, rr = col & 255;
  int ky = rr >> 4, kx = rr & 15;
  long long src = (((long long)(b * 3 + c) * 224) + py * 16 + ky) * 224 + px * 16 + kx;
  Ap[idx] = f2bf(fov[src]);
}

__global__ __launch_bounds__(256) void f32_to_bf16_k(
    const float* __restrict__ in, __bf16* __restrict__ out, long long n)
{
  long long i = (long long)blockIdx.x * 256 + threadIdx.x;
  if (i < n) out[i] = f2bf(in[i]);
}

__global__ void head_concat(const float* __restrict__ cur, const float* __restrict__ tgt,
                            __bf16* __restrict__ Ah)
{
  int i = threadIdx.x;             // 128 threads: b = i/4, j = i%4
  if (i < 128) {
    int b = i >> 2, j = i & 3;
    float v = (j < 2) ? cur[b * 2 + j] : tgt[b * 2 + (j - 2)];
    Ah[(long long)b * 544 + j] = f2bf(v);
  }
}

__global__ __launch_bounds__(256) void reduce_splitk(
    const float* __restrict__ part, float* __restrict__ outp,
    int n, int parts, long long stride)
{
  int i = blockIdx.x * 256 + threadIdx.x;
  if (i >= n) return;
  float s = 0.f;
  for (int z = 0; z < parts; ++z) s += part[(long long)z * stride + i];
  outp[i] = s;
}

extern "C" void kernel_launch(void* const* d_in, const int* in_sizes, int n_in,
                              void* d_out, int out_size, void* d_ws, size_t ws_size,
                              hipStream_t stream)
{
  const float* fov    = (const float*)d_in[0];
  const float* curp   = (const float*)d_in[1];
  const float* tgtp   = (const float*)d_in[2];
  const float* conv_w = (const float*)d_in[3];
  const float* conv_b = (const float*)d_in[4];
  const float* lnin_g = (const float*)d_in[5];
  const float* lnin_b = (const float*)d_in[6];
  const float* wq     = (const float*)d_in[7];
  const float* wk     = (const float*)d_in[8];
  const float* wvw    = (const float*)d_in[9];
  const float* ln1_g  = (const float*)d_in[10];
  const float* ln1_b  = (const float*)d_in[11];
  const float* w1     = (const float*)d_in[12];
  const float* b1     = (const float*)d_in[13];
  const float* ln2_g  = (const float*)d_in[14];
  const float* ln2_b  = (const float*)d_in[15];
  const float* w2     = (const float*)d_in[16];
  const float* b2     = (const float*)d_in[17];
  const float* ln3_g  = (const float*)d_in[18];
  const float* ln3_b  = (const float*)d_in[19];
  const float* img_g  = (const float*)d_in[20];
  const float* img_b  = (const float*)d_in[21];
  const float* t1w    = (const float*)d_in[22];
  const float* t1b    = (const float*)d_in[23];
  const float* tl1g   = (const float*)d_in[24];
  const float* tl1b   = (const float*)d_in[25];
  const float* t2w    = (const float*)d_in[26];
  const float* t2b    = (const float*)d_in[27];
  const float* tl2g   = (const float*)d_in[28];
  const float* tl2b   = (const float*)d_in[29];
  const float* h1w    = (const float*)d_in[30];
  const float* h1b    = (const float*)d_in[31];
  const float* hlg    = (const float*)d_in[32];
  const float* hlb    = (const float*)d_in[33];
  const float* h2w    = (const float*)d_in[34];
  const float* h2b    = (const float*)d_in[35];

  // ---- workspace arena (~203 MB) ----
  char* ws = (char*)d_ws;
  size_t off = 0;
  auto alloc = [&](size_t bytes) {
    void* p = ws + off;
    off = (off + bytes + 255) & ~(size_t)255;
    return p;
  };
  float*  x    = (float*)alloc((size_t)ROWS * EMB * 4);      // residual stream
  __bf16* xn   = (__bf16*)alloc((size_t)ROWS * EMB * 2);     // LN/bf16 activations (also patch-A, fbf)
  float*  q    = (float*)alloc((size_t)ROWS * EMB * 4);
  float*  k    = (float*)alloc((size_t)ROWS * EMB * 4);
  float*  v    = (float*)alloc((size_t)ROWS * EMB * 4);
  __bf16* qbf  = (__bf16*)alloc((size_t)ROWS * EMB * 2);
  __bf16* kbf  = (__bf16*)alloc((size_t)ROWS * EMB * 2);
  size_t  sBytes = (size_t)BATCH * NHEAD * TOK * LD_S * 4;   // 62.6 MB
  size_t  pBytes = (size_t)BATCH * NHEAD * TOK * LD_P * 2;   // 33.7 MB
  float*  S    = (float*)alloc(sBytes + pBytes);             // att region (reused as h1)
  __bf16* P    = (__bf16*)((char*)S + sBytes);
  float*  h1   = S;                                          // [6272,3072] f32 (77MB <= 96MB)
  __bf16* h1n  = (__bf16*)q;                                 // [6272,3072] bf16 == q+k bytes
  float*  h2   = v;                                          // [6272,768] f32
  (void)ws_size; (void)in_sizes; (void)n_in; (void)out_size;

  const long long nEl = (long long)ROWS * EMB;
  const int cvtGrid = (int)((nEl + 255) / 256);

  // ---- patch embed: im2col + A x conv_w^T + conv_b ----
  im2col_bf16<<<cvtGrid, 256, 0, stream>>>(fov, xn);
  {
    dim3 g(6, 49, 1);
    gemm_wmma_bf16<true, false, false, false><<<g, 256, 0, stream>>>(
        xn, conv_w, x, conv_b, ROWS, EMB, 768, 768, 768, EMB,
        0, 0, 0, 0, 0, 0, 1, 0);
  }

  // ---- transformer blocks ----
  for (int i = 0; i < NBLK; ++i) {
    const long long oE = (long long)i * EMB;
    ln_rows<0><<<ROWS, 256, 0, stream>>>(x, nullptr, lnin_g + oE, lnin_b + oE,
                                         xn, EMB, EMB, EMB);
    dim3 gq(6, 49, 1);
    gemm_wmma_bf16<false, false, false, false><<<gq, 256, 0, stream>>>(
        xn, wq + (long long)i * EMB * EMB, q, nullptr,
        ROWS, EMB, EMB, EMB, EMB, EMB, 0, 0, 0, 0, 0, 0, 1, 0);
    gemm_wmma_bf16<false, false, false, false><<<gq, 256, 0, stream>>>(
        xn, wk + (long long)i * EMB * EMB, k, nullptr,
        ROWS, EMB, EMB, EMB, EMB, EMB, 0, 0, 0, 0, 0, 0, 1, 0);
    gemm_wmma_bf16<false, false, false, false><<<gq, 256, 0, stream>>>(
        xn, wvw + (long long)i * EMB * EMB, v, nullptr,
        ROWS, EMB, EMB, EMB, EMB, EMB, 0, 0, 0, 0, 0, 0, 1, 0);
    f32_to_bf16_k<<<cvtGrid, 256, 0, stream>>>(q, qbf, nEl);
    f32_to_bf16_k<<<cvtGrid, 256, 0, stream>>>(k, kbf, nEl);

    // S[b,h] = Q_bh K_bh^T  (batched, B^T bf16 path, async staging)
    dim3 gs(2, 2, BATCH * NHEAD);
    gemm_wmma_bf16<true, true, false, false><<<gs, 256, 0, stream>>>(
        qbf, kbf, S, nullptr, TOK, TOK, DH, EMB, EMB, LD_S,
        (long long)TOK * EMB, DH, (long long)TOK * EMB, DH,
        (long long)NHEAD * TOK * LD_S, (long long)TOK * LD_S, NHEAD, 0);

    softmax_bf16<<<BATCH * NHEAD * TOK, 256, 0, stream>>>(
        S, P, TOK, LD_S, LD_P, 1.0f / 14.0f);   // scale = 1/sqrt(T)

    // x += P @ V  (batched, accumulate epilogue)
    dim3 go(1, 2, BATCH * NHEAD);
    gemm_wmma_bf16<false, false, true, false><<<go, 256, 0, stream>>>(
        P, v, x, nullptr, TOK, DH, TOK, LD_P, EMB, EMB,
        (long long)NHEAD * TOK * LD_P, (long long)TOK * LD_P,
        (long long)TOK * EMB, DH,
        (long long)TOK * EMB, DH, NHEAD, 0);

    // MLP: x += LN3( relu(LN2(LN1(x)@w1 + b1)) @ w2 + b2 )
    ln_rows<0><<<ROWS, 256, 0, stream>>>(x, nullptr, ln1_g + oE, ln1_b + oE,
                                         xn, EMB, EMB, EMB);
    dim3 g1(24, 49, 1);
    gemm_wmma_bf16<false, false, false, false><<<g1, 256, 0, stream>>>(
        xn, w1 + (long long)i * EMB * FFDIM, h1, b1 + (long long)i * FFDIM,
        ROWS, FFDIM, EMB, EMB, FFDIM, FFDIM, 0, 0, 0, 0, 0, 0, 1, 0);
    ln_rows<1><<<ROWS, 256, 0, stream>>>(h1, nullptr,
                                         ln2_g + (long long)i * FFDIM,
                                         ln2_b + (long long)i * FFDIM,
                                         h1n, FFDIM, FFDIM, FFDIM);
    dim3 g2(6, 49, 1);
    gemm_wmma_bf16<false, false, false, false><<<g2, 256, 0, stream>>>(
        h1n, w2 + (long long)i * FFDIM * EMB, h2, b2 + oE,
        ROWS, EMB, FFDIM, FFDIM, EMB, EMB, 0, 0, 0, 0, 0, 0, 1, 0);
    ln_rows<2><<<ROWS, 256, 0, stream>>>(h2, nullptr, ln3_g + oE, ln3_b + oE,
                                         x, EMB, EMB, EMB);
  }

  // ---- trunk ----
  __bf16* fbf  = xn;                 // [32, 150528] bf16 (same byte size as xn)
  float*  part = (float*)P;          // split-K partials: 64 x [32,1024] = 8 MB
  float*  t1   = q;
  __bf16* f2   = (__bf16*)k;
  float*  t2   = v;
  __bf16* Ahd  = (__bf16*)S;         // [32,544] bf16 head-A (+slack)
  float*  t3   = (float*)qbf;
  __bf16* hbf  = kbf;

  ln_rows<0><<<BATCH, 256, 0, stream>>>(x, nullptr, img_g, img_b,
                                        fbf, (int)FLATD, (int)FLATD, (int)FLATD);
  {
    // weights read exactly once: split-K across 64 chunks, deterministic reduce
    dim3 gt(8, 1, 64);
    gemm_wmma_bf16<false, false, false, true><<<gt, 256, 0, stream>>>(
        fbf, t1w, part, nullptr, BATCH, 1024, (int)FLATD, (int)FLATD, 1024, 1024,
        0, 0, 0, 0, (long long)BATCH * 1024, 0, 1, 2368);
    reduce_splitk<<<(BATCH * 1024 + 255) / 256, 256, 0, stream>>>(
        part, t1, BATCH * 1024, 64, (long long)BATCH * 1024);
  }
  ln_rows<1><<<BATCH, 256, 0, stream>>>(t1, t1b, tl1g, tl1b, f2, 1024, 1024, 1024);
  {
    dim3 g(4, 1, 1);
    gemm_wmma_bf16<false, false, false, false><<<g, 256, 0, stream>>>(
        f2, t2w, t2, t2b, BATCH, TFD, 1024, 1024, TFD, TFD,
        0, 0, 0, 0, 0, 0, 1, 0);
  }
  // concat([cur, tgt, f]) as bf16 head-A with ld 544 (zero pad cols 516..543)
  ln_rows<1><<<BATCH, 256, 0, stream>>>(t2, nullptr, tl2g, tl2b,
                                        (void*)(Ahd + 4), TFD, TFD, 544);
  head_concat<<<1, 128, 0, stream>>>(curp, tgtp, Ahd);
  {
    dim3 g(4, 1, 1);
    gemm_wmma_bf16<false, false, false, false><<<g, 256, 0, stream>>>(
        Ahd, h1w, t3, h1b, BATCH, TFD, TFD + 4, 544, TFD, TFD,
        0, 0, 0, 0, 0, 0, 1, 0);
  }
  ln_rows<1><<<BATCH, 256, 0, stream>>>(t3, nullptr, hlg, hlb, hbf, TFD, TFD, TFD);
  {
    dim3 g(1, 1, 1);
    gemm_wmma_bf16<false, false, false, false><<<g, 256, 0, stream>>>(
        hbf, h2w, (float*)d_out, h2b, BATCH, 4, TFD, TFD, 4, 4,
        0, 0, 0, 0, 0, 0, 1, 0);
  }
}
